// StateEncoder_10256381903590
// MI455X (gfx1250) — compile-verified, matrix operand order checked by reading, not run on hardware
//
#include <hip/hip_runtime.h>

#define NNODES 50000
#define NEDGES 800000
#define ETOT   (NEDGES + NNODES)
#define NGRAPH 64
#define LN_EPS 1e-5f

typedef __attribute__((ext_vector_type(16))) __bf16 v16bf;
typedef __attribute__((ext_vector_type(8)))  __bf16 v8bf;
typedef __attribute__((ext_vector_type(8)))  float  v8f;

// ---------------- conversion kernels ----------------

__global__ void k_f32_to_bf16(const float* __restrict__ in, __bf16* __restrict__ out, int n) {
    int i = blockIdx.x * blockDim.x + threadIdx.x;
    if (i < n) out[i] = (__bf16)in[i];
}

// W [K, F] f32 row-major -> Wt [F, K] bf16 (transposed so B-fragment loads are contiguous)
__global__ void k_transpose_w_bf16(const float* __restrict__ W, __bf16* __restrict__ Wt,
                                   int K, int F) {
    int i = blockIdx.x * blockDim.x + threadIdx.x;
    if (i < K * F) {
        int k = i / F, n = i % F;
        Wt[n * K + k] = (__bf16)W[i];
    }
}

__global__ void k_fill_f32(float* __restrict__ p, float v, int n) {
    int i = blockIdx.x * blockDim.x + threadIdx.x;
    if (i < n) p[i] = v;
}

// ---------------- WMMA GEMM: C[M,F] = A[M,K] * B[K,F] (bf16 in, f32 out) ----------------
// 1 wave -> one 16x16 tile. block = 128 threads (4 waves) covering 16 rows x 64 cols.

__global__ void k_gemm_wmma_bf16(const __bf16* __restrict__ A,   // [M,K] row-major
                                 const __bf16* __restrict__ Bt,  // [F,K] (W transposed)
                                 float* __restrict__ C, int M, int K, int F) {
    const int wave = threadIdx.x >> 5;
    const int lane = threadIdx.x & 31;
    const int tileM = blockIdx.x * 16;
    const int tileN = (blockIdx.y * 4 + wave) * 16;
    if (tileN >= F) return;

    const int half = lane >> 4;   // 0: K base +0, 1: K base +8 (per ISA A/B fragment layout)
    const int l16  = lane & 15;
    const int arow = tileM + l16;
    const int bcol = tileN + l16;

    v8f acc = {};
    for (int k = 0; k < K; k += 32) {
        const int kb = k + half * 8;
        v8bf a0 = *(const v8bf*)(A  + (size_t)arow * K + kb);
        v8bf a1 = *(const v8bf*)(A  + (size_t)arow * K + kb + 16);
        v8bf b0 = *(const v8bf*)(Bt + (size_t)bcol * K + kb);
        v8bf b1 = *(const v8bf*)(Bt + (size_t)bcol * K + kb + 16);
        v16bf av, bv;
#pragma unroll
        for (int i = 0; i < 8; ++i) {
            av[i] = a0[i]; av[8 + i] = a1[i];
            bv[i] = b0[i]; bv[8 + i] = b1[i];
        }
        acc = __builtin_amdgcn_wmma_f32_16x16x32_bf16(
            /*neg_a=*/false, av, /*neg_b=*/false, bv,
            /*c_mod=*/(short)0, acc, /*reuse_a=*/false, /*reuse_b=*/false);
    }
    // D layout: col = lane%16, row = vgpr + 8*(lane/16)
    const int orow = tileM + half * 8;
#pragma unroll
    for (int v = 0; v < 8; ++v)
        C[(size_t)(orow + v) * F + tileN + l16] = acc[v];
}

// ---------------- attention coefficients a_l / a_r ----------------

__global__ void k_attn_coef(const float* __restrict__ xl, const float* __restrict__ asrc,
                            const float* __restrict__ adst, float* __restrict__ al,
                            float* __restrict__ ar, int H, int Cc) {
    int idx = blockIdx.x * blockDim.x + threadIdx.x;  // node*H + h
    if (idx >= NNODES * H) return;
    int n = idx / H, h = idx % H;
    const float* row = xl + (size_t)n * H * Cc + (size_t)h * Cc;
    float sl = 0.f, sr = 0.f;
    for (int c = 0; c < Cc; ++c) {
        float v = row[c];
        sl += v * asrc[h * Cc + c];
        sr += v * adst[h * Cc + c];
    }
    al[idx] = sl;
    ar[idx] = sr;
}

// ---------------- edge softmax (3 passes) + aggregation ----------------

__device__ __forceinline__ void edge_sd(const int* __restrict__ ei, int e, int& s, int& d) {
    if (e < NEDGES) { s = ei[e]; d = ei[NEDGES + e]; }
    else            { s = d = e - NEDGES; }            // self loops appended
}

__device__ __forceinline__ void atomicMaxF(float* addr, float v) {
    // sign-aware bit-order punning: correct ordering for mixed-sign floats
    if (v >= 0.f) atomicMax((int*)addr, __float_as_int(v));
    else          atomicMin((unsigned int*)addr, __float_as_uint(v));
}

__device__ __forceinline__ float leaky02(float x) { return x > 0.f ? x : 0.2f * x; }

__global__ void k_edge_max(const int* __restrict__ ei, const float* __restrict__ al,
                           const float* __restrict__ ar, float* __restrict__ m, int H) {
    int idx = blockIdx.x * blockDim.x + threadIdx.x;
    if (idx >= ETOT * H) return;
    int e = idx / H, h = idx % H, s, d;
    edge_sd(ei, e, s, d);
    atomicMaxF(&m[d * H + h], leaky02(al[s * H + h] + ar[d * H + h]));
}

__global__ void k_edge_expsum(const int* __restrict__ ei, const float* __restrict__ al,
                              const float* __restrict__ ar, const float* __restrict__ m,
                              float* __restrict__ ssum, float* __restrict__ alpha, int H) {
    int idx = blockIdx.x * blockDim.x + threadIdx.x;
    if (idx >= ETOT * H) return;
    int e = idx / H, h = idx % H, s, d;
    edge_sd(ei, e, s, d);
    float ev = leaky02(al[s * H + h] + ar[d * H + h]);
    float p = __expf(ev - m[d * H + h]);
    alpha[idx] = p;
    atomicAdd(&ssum[d * H + h], p);
}

__global__ void k_edge_norm(const int* __restrict__ ei, const float* __restrict__ ssum,
                            float* __restrict__ alpha, int H) {
    int idx = blockIdx.x * blockDim.x + threadIdx.x;
    if (idx >= ETOT * H) return;
    int e = idx / H, h = idx % H, s, d;
    edge_sd(ei, e, s, d);
    alpha[idx] = alpha[idx] / (ssum[d * H + h] + 1e-16f);
}

// wave per edge; channels strided across 32 lanes; L2-resident atomics
__global__ void k_edge_aggr(const int* __restrict__ ei, const float* __restrict__ alpha,
                            const float* __restrict__ xl, float* __restrict__ out,
                            int H, int Cc) {
    int wv = (blockIdx.x * blockDim.x + threadIdx.x) >> 5;
    int lane = threadIdx.x & 31;
    if (wv >= ETOT) return;
    int s, d;
    edge_sd(ei, wv, s, d);
    const int F = H * Cc;
    for (int c = lane; c < F; c += 32) {
        float a = alpha[(size_t)wv * H + (c / Cc)];
        atomicAdd(&out[(size_t)d * F + c], a * xl[(size_t)s * F + c]);
    }
}

// ---------------- bias + LayerNorm + ReLU (wave per node) ----------------

__global__ void k_ln_relu(const float* __restrict__ agg, const float* __restrict__ bias,
                          const float* __restrict__ g, const float* __restrict__ be,
                          float* __restrict__ out, int F) {
    int node = (blockIdx.x * blockDim.x + threadIdx.x) >> 5;
    int lane = threadIdx.x & 31;
    if (node >= NNODES) return;
    const float* row = agg + (size_t)node * F;
    const int nv = F >> 5;   // 4 or 8
    float vals[8];
    float sum = 0.f, sq = 0.f;
    for (int i = 0; i < nv; ++i) {
        int c = lane + 32 * i;
        float v = row[c] + bias[c];
        vals[i] = v; sum += v; sq += v * v;
    }
#pragma unroll
    for (int o = 16; o > 0; o >>= 1) {
        sum += __shfl_xor(sum, o, 32);
        sq  += __shfl_xor(sq,  o, 32);
    }
    float mu  = sum / (float)F;
    float var = sq / (float)F - mu * mu;
    float inv = rsqrtf(var + LN_EPS);
    for (int i = 0; i < nv; ++i) {
        int c = lane + 32 * i;
        float v = (vals[i] - mu) * inv * g[c] + be[c];
        out[(size_t)node * F + c] = v > 0.f ? v : 0.f;
    }
}

// ---------------- pooling + head ----------------

__global__ void k_pool_init(float* __restrict__ sum, float* __restrict__ mx,
                            float* __restrict__ cnt) {
    int i = blockIdx.x * blockDim.x + threadIdx.x;
    if (i < NGRAPH * 128) { sum[i] = 0.f; mx[i] = -__builtin_inff(); }
    if (i < NGRAPH) cnt[i] = 0.f;
}

__global__ void k_pool_accum(const float* __restrict__ h, const int* __restrict__ batch,
                             float* __restrict__ sum, float* __restrict__ mx,
                             float* __restrict__ cnt) {
    int node = (blockIdx.x * blockDim.x + threadIdx.x) >> 5;
    int lane = threadIdx.x & 31;
    if (node >= NNODES) return;
    int b = batch[node];
    for (int c = lane; c < 128; c += 32) {
        float v = h[(size_t)node * 128 + c];
        atomicAdd(&sum[b * 128 + c], v);
        atomicMaxF(&mx[b * 128 + c], v);
    }
    if (lane == 0) atomicAdd(&cnt[b], 1.f);
}

__global__ void k_final_proj(const float* __restrict__ sum, const float* __restrict__ mx,
                             const float* __restrict__ cnt, const float* __restrict__ Wp,
                             const float* __restrict__ bp, float* __restrict__ out) {
    int idx = blockIdx.x * blockDim.x + threadIdx.x;
    if (idx >= NGRAPH * 128) return;
    int gph = idx / 128, j = idx % 128;
    float inv = 1.f / fmaxf(cnt[gph], 1.f);
    float acc = bp[j];
    for (int k = 0; k < 128; ++k) acc += (sum[gph * 128 + k] * inv) * Wp[k * 128 + j];
    for (int k = 0; k < 128; ++k) acc += mx[gph * 128 + k] * Wp[(128 + k) * 128 + j];
    out[idx] = acc;
}

// ---------------- host orchestration ----------------

static inline int ceil_div(int a, int b) { return (a + b - 1) / b; }

static void run_gat_layer(const float* act_in, int K,
                          const float* W, const float* asrc, const float* adst,
                          const float* bias, const float* gm, const float* be,
                          int H, int Cc,
                          float* xl, float* agg, float* hout,
                          __bf16* actbf, __bf16* wtbf,
                          float* al, float* ar, float* m, float* s, float* alpha,
                          const int* ei, hipStream_t stream) {
    const int F = H * Cc;
    // convert activations + weights to bf16 (weights transposed)
    k_f32_to_bf16<<<ceil_div(NNODES * K, 256), 256, 0, stream>>>(act_in, actbf, NNODES * K);
    k_transpose_w_bf16<<<ceil_div(K * F, 256), 256, 0, stream>>>(W, wtbf, K, F);
    // xl = act @ W   (WMMA)
    dim3 ggrid(NNODES / 16, F / 64);
    k_gemm_wmma_bf16<<<ggrid, 128, 0, stream>>>(actbf, wtbf, xl, NNODES, K, F);
    // attention coefficients
    k_attn_coef<<<ceil_div(NNODES * H, 256), 256, 0, stream>>>(xl, asrc, adst, al, ar, H, Cc);
    // init softmax state + accumulator
    k_fill_f32<<<ceil_div(NNODES * H, 256), 256, 0, stream>>>(m, -__builtin_inff(), NNODES * H);
    k_fill_f32<<<ceil_div(NNODES * H, 256), 256, 0, stream>>>(s, 0.f, NNODES * H);
    k_fill_f32<<<ceil_div(NNODES * F, 256), 256, 0, stream>>>(agg, 0.f, NNODES * F);
    // segment softmax over dst
    k_edge_max   <<<ceil_div(ETOT * H, 256), 256, 0, stream>>>(ei, al, ar, m, H);
    k_edge_expsum<<<ceil_div(ETOT * H, 256), 256, 0, stream>>>(ei, al, ar, m, s, alpha, H);
    k_edge_norm  <<<ceil_div(ETOT * H, 256), 256, 0, stream>>>(ei, s, alpha, H);
    // weighted aggregation (wave per edge)
    k_edge_aggr<<<ceil_div(ETOT * 32, 256), 256, 0, stream>>>(ei, alpha, xl, agg, H, Cc);
    // bias + LN + ReLU (wave per node)
    k_ln_relu<<<ceil_div(NNODES * 32, 256), 256, 0, stream>>>(agg, bias, gm, be, hout, F);
}

extern "C" void kernel_launch(void* const* d_in, const int* in_sizes, int n_in,
                              void* d_out, int out_size, void* d_ws, size_t ws_size,
                              hipStream_t stream) {
    const float* x     = (const float*)d_in[0];
    const int*   ei    = (const int*)  d_in[1];
    const int*   batch = (const int*)  d_in[2];
    const float* W1 = (const float*)d_in[3];
    const float* as1 = (const float*)d_in[4];
    const float* ad1 = (const float*)d_in[5];
    const float* b1 = (const float*)d_in[6];
    const float* g1 = (const float*)d_in[7];
    const float* be1 = (const float*)d_in[8];
    const float* W2 = (const float*)d_in[9];
    const float* as2 = (const float*)d_in[10];
    const float* ad2 = (const float*)d_in[11];
    const float* b2 = (const float*)d_in[12];
    const float* g2 = (const float*)d_in[13];
    const float* be2 = (const float*)d_in[14];
    const float* W3 = (const float*)d_in[15];
    const float* as3 = (const float*)d_in[16];
    const float* ad3 = (const float*)d_in[17];
    const float* b3 = (const float*)d_in[18];
    const float* g3 = (const float*)d_in[19];
    const float* be3 = (const float*)d_in[20];
    const float* Wp = (const float*)d_in[21];
    const float* bp = (const float*)d_in[22];

    // workspace carve-up
    size_t off = 0;
    auto carve = [&](size_t bytes) -> void* {
        void* p = (char*)d_ws + off;
        off += (bytes + 255) & ~(size_t)255;
        return p;
    };
    float*  xlbuf = (float*) carve((size_t)NNODES * 256 * 4);
    float*  aggbf = (float*) carve((size_t)NNODES * 256 * 4);
    float*  hbuf  = (float*) carve((size_t)NNODES * 256 * 4);
    __bf16* actbf = (__bf16*)carve((size_t)NNODES * 256 * 2);
    __bf16* wtbf  = (__bf16*)carve((size_t)256 * 256 * 2);
    float*  al    = (float*) carve((size_t)NNODES * 4 * 4);
    float*  ar    = (float*) carve((size_t)NNODES * 4 * 4);
    float*  mbuf  = (float*) carve((size_t)NNODES * 4 * 4);
    float*  sbuf  = (float*) carve((size_t)NNODES * 4 * 4);
    float*  alpha = (float*) carve((size_t)ETOT * 4 * 4);
    float*  psum  = (float*) carve((size_t)NGRAPH * 128 * 4);
    float*  pmax  = (float*) carve((size_t)NGRAPH * 128 * 4);
    float*  pcnt  = (float*) carve((size_t)NGRAPH * 4);

    // Layer 1: [N,128] -> [N,256], H=4, C=64
    run_gat_layer(x,    128, W1, as1, ad1, b1, g1, be1, 4, 64,
                  xlbuf, aggbf, hbuf, actbf, wtbf, al, ar, mbuf, sbuf, alpha, ei, stream);
    // Layer 2: [N,256] -> [N,256], H=4, C=64
    run_gat_layer(hbuf, 256, W2, as2, ad2, b2, g2, be2, 4, 64,
                  xlbuf, aggbf, hbuf, actbf, wtbf, al, ar, mbuf, sbuf, alpha, ei, stream);
    // Layer 3: [N,256] -> [N,128], H=1, C=128 (mean over 1 head == identity)
    run_gat_layer(hbuf, 256, W3, as3, ad3, b3, g3, be3, 1, 128,
                  xlbuf, aggbf, hbuf, actbf, wtbf, al, ar, mbuf, sbuf, alpha, ei, stream);

    // pooling (mean + max per graph) + projection head
    k_pool_init<<<ceil_div(NGRAPH * 128, 256), 256, 0, stream>>>(psum, pmax, pcnt);
    k_pool_accum<<<ceil_div(NNODES * 32, 256), 256, 0, stream>>>(hbuf, batch, psum, pmax, pcnt);
    k_final_proj<<<ceil_div(NGRAPH * 128, 256), 256, 0, stream>>>(psum, pmax, pcnt, Wp, bp,
                                                                  (float*)d_out);
}
